// Net_61014305406978
// MI455X (gfx1250) — compile-verified, matrix operand order checked by reading, not run on hardware
//
#include <hip/hip_runtime.h>
#include <hip/hip_bf16.h>
#include <float.h>
#include <stdint.h>

typedef __attribute__((ext_vector_type(16))) _Float16 v16h;
typedef __attribute__((ext_vector_type(8)))  _Float16 v8h;
typedef __attribute__((ext_vector_type(8)))  float    v8f;

// ---------------------------------------------------------------------------
// WMMA helpers (CDNA5 gfx1250, wave32): D(16x16,f32) = A(16x32,f16) x B(32x16,f16) + C
// ---------------------------------------------------------------------------
__device__ __forceinline__ v8f wmma32(v16h a, v16h b, v8f c) {
    return __builtin_amdgcn_wmma_f32_16x16x32_f16(
        /*neg_a=*/false, a, /*neg_b=*/false, b,
        /*c_mod=*/(short)0, c, /*reuse_a=*/false, /*reuse_b=*/false);
}

// Load one 16-element f16 fragment for a lane. `p` must point at
// (row base + kc + hsel*8) and be 16-byte aligned. Elements i=0..7 -> k offset
// +0..+7, i=8..15 -> k offset +16..+23 (ISA 16-bit A/B layout).
__device__ __forceinline__ v16h load_frag(const _Float16* p) {
    v8h lo = *(const v8h*)(p);
    v8h hi = *(const v8h*)(p + 16);
    v16h a;
#pragma unroll
    for (int i = 0; i < 8; ++i) { a[i] = lo[i]; a[i + 8] = hi[i]; }
    return a;
}

__device__ __forceinline__ void atomicMaxF32(float* addr, float val) {
    int* ia = (int*)addr;
    int cur = __float_as_int(*addr);
    while (val > __int_as_float(cur)) {
        int old = atomicCAS(ia, cur, __float_as_int(val));
        if (old == cur) break;
        cur = old;
    }
}

__device__ __forceinline__ float bn_rs() { return rsqrtf(1.0f + 1e-5f); }

// argmax combine with reference tie-breaking (first/lowest index wins ties)
__device__ __forceinline__ void argmax_combine(float& v, int& i, float v2, int i2) {
    if (v2 > v || (v2 == v && i2 < i)) { v = v2; i = i2; }
}

// ---------------------------------------------------------------------------
// Farthest point sampling: one block (1024 threads = 32 waves) per cloud.
// Matches the reference scan: idx[0]=0; idx[t]=argmax(min_d) (first-max ties).
// Reduction: 5 shfl steps in-wave + 32-partial wave-0 reduce -> 2 barriers/iter.
// ---------------------------------------------------------------------------
__global__ void __launch_bounds__(1024) fps_kernel(const float* __restrict__ pos,
                                                   int* __restrict__ outIdx,
                                                   int N, int nOut) {
    __shared__ float px[4096], py[4096], pz[4096], mind[4096];
    __shared__ float rv[32];
    __shared__ int   ri[32];
    __shared__ int   lastSh;
    const int b = blockIdx.x;
    const int tid = threadIdx.x;
    const int wave = tid >> 5;
    const int lane = tid & 31;
    for (int i = tid; i < N; i += 1024) {
        px[i] = pos[(b * N + i) * 3 + 0];
        py[i] = pos[(b * N + i) * 3 + 1];
        pz[i] = pos[(b * N + i) * 3 + 2];
        mind[i] = 1e10f;
    }
    __syncthreads();
    int last = 0;
    for (int t = 0; t < nOut; ++t) {
        if (tid == 0) outIdx[b * nOut + t] = last;
        const float lx = px[last], ly = py[last], lz = pz[last];
        float bv = -1.0f; int bi = 0;
        for (int i = tid; i < N; i += 1024) {
            float dx = px[i] - lx, dy = py[i] - ly, dz = pz[i] - lz;
            float md = fminf(mind[i], dx * dx + dy * dy + dz * dz);
            mind[i] = md;
            if (md > bv) { bv = md; bi = i; }
        }
        // in-wave argmax (wave32)
#pragma unroll
        for (int s = 16; s > 0; s >>= 1) {
            float v2 = __shfl_xor(bv, s, 32);
            int   i2 = __shfl_xor(bi, s, 32);
            argmax_combine(bv, bi, v2, i2);
        }
        if (lane == 0) { rv[wave] = bv; ri[wave] = bi; }
        __syncthreads();
        if (wave == 0) {
            float v = rv[lane]; int i = ri[lane];
#pragma unroll
            for (int s = 16; s > 0; s >>= 1) {
                float v2 = __shfl_xor(v, s, 32);
                int   i2 = __shfl_xor(i, s, 32);
                argmax_combine(v, i, v2, i2);
            }
            if (lane == 0) lastSh = i;
        }
        __syncthreads();
        last = lastSh;
    }
}

// ---------------------------------------------------------------------------
// Gather sampled center coordinates.
// ---------------------------------------------------------------------------
__global__ void gather_centers_kernel(const float* __restrict__ pos, const int* __restrict__ idx,
                                      float* __restrict__ cen, int N, int M, int total) {
    int t = blockIdx.x * 256 + threadIdx.x;
    if (t >= total) return;
    int b = t / M;
    int j = idx[t];
    cen[t * 3 + 0] = pos[(b * N + j) * 3 + 0];
    cen[t * 3 + 1] = pos[(b * N + j) * 3 + 1];
    cen[t * 3 + 2] = pos[(b * N + j) * 3 + 2];
}

// ---------------------------------------------------------------------------
// Radius neighbors: thread per center keeps the 64 nearest points within r.
// (Order-free; the downstream max-aggregation is permutation invariant.)
// ---------------------------------------------------------------------------
__global__ void radius_kernel(const float* __restrict__ pos, const float* __restrict__ cen,
                              int* __restrict__ nidx, int* __restrict__ cnt,
                              float* __restrict__ d2buf, int N, int M, int total, float r2) {
    int t = blockIdx.x * 256 + threadIdx.x;
    if (t >= total) return;
    int b = t / M;
    const float cx = cen[t * 3 + 0], cy = cen[t * 3 + 1], cz = cen[t * 3 + 2];
    const int base = t * 64;
    int c = 0; float worst = -1.0f; int wslot = 0;
    for (int i = 0; i < N; ++i) {
        float dx = pos[(b * N + i) * 3 + 0] - cx;
        float dy = pos[(b * N + i) * 3 + 1] - cy;
        float dz = pos[(b * N + i) * 3 + 2] - cz;
        float d2 = dx * dx + dy * dy + dz * dz;
        if (d2 <= r2) {
            if (c < 64) {
                nidx[base + c] = i; d2buf[base + c] = d2;
                if (d2 > worst) { worst = d2; wslot = c; }
                ++c;
            } else if (d2 < worst) {
                nidx[base + wslot] = i; d2buf[base + wslot] = d2;
                worst = -1.0f;
                for (int j = 0; j < 64; ++j) {
                    float v = d2buf[base + j];
                    if (v > worst) { worst = v; wslot = j; }
                }
            }
        }
    }
    cnt[t] = c;
}

// ---------------------------------------------------------------------------
// One MLP layer inside the fused PointConv block (LDS -> LDS, f16, WMMA).
// in/out strides are 264 halfs (16B aligned rows). Wt is [cout][cinPad] f16.
// ---------------------------------------------------------------------------
__device__ __forceinline__ void pc_layer(const _Float16* __restrict__ in, _Float16* __restrict__ out,
                                         const _Float16* __restrict__ Wt,
                                         const float* __restrict__ bias,
                                         const float* __restrict__ gamma,
                                         const float* __restrict__ beta,
                                         int cinPad, int cout) {
    const int lane = threadIdx.x & 31;
    const int wave = threadIdx.x >> 5;
    const int nlo = lane & 15;
    const int hsel = (lane >> 4) & 1;
    const int ntiles = cout >> 4;
    for (int tile = wave; tile < 4 * ntiles; tile += 8) {
        const int mt = tile & 3;
        const int nt = tile >> 2;
        const int col = nt * 16 + nlo;
        v8f acc = {};
        for (int kc = 0; kc < cinPad; kc += 32) {
            __builtin_prefetch(Wt + col * cinPad + kc + 32 + hsel * 8, 0, 1);
            v16h bf = load_frag(Wt + col * cinPad + kc + hsel * 8);
            v16h af = load_frag(in + (mt * 16 + nlo) * 264 + kc + hsel * 8);
            acc = wmma32(af, bf, acc);
        }
        const float bs = bias[col];
        const float gs = gamma[col] * bn_rs();
        const float bt = beta[col];
#pragma unroll
        for (int v = 0; v < 8; ++v) {
            int row = mt * 16 + v + hsel * 8;
            float y = fmaxf(acc[v] + bs, 0.0f) * gs + bt;
            out[row * 264 + col] = (_Float16)y;
        }
    }
}

// Final layer: fuse masked column-max (valid rows only) via shfl + shared CAS-max.
__device__ __forceinline__ void pc_layer_final(const _Float16* __restrict__ in,
                                               const _Float16* __restrict__ Wt,
                                               const float* __restrict__ bias,
                                               const float* __restrict__ gamma,
                                               const float* __restrict__ beta,
                                               float* __restrict__ colmax,
                                               int myCnt, int cinPad, int cout) {
    const int lane = threadIdx.x & 31;
    const int wave = threadIdx.x >> 5;
    const int nlo = lane & 15;
    const int hsel = (lane >> 4) & 1;
    const int ntiles = cout >> 4;
    for (int tile = wave; tile < 4 * ntiles; tile += 8) {
        const int mt = tile & 3;
        const int nt = tile >> 2;
        const int col = nt * 16 + nlo;
        v8f acc = {};
        for (int kc = 0; kc < cinPad; kc += 32) {
            __builtin_prefetch(Wt + col * cinPad + kc + 32 + hsel * 8, 0, 1);
            v16h bf = load_frag(Wt + col * cinPad + kc + hsel * 8);
            v16h af = load_frag(in + (mt * 16 + nlo) * 264 + kc + hsel * 8);
            acc = wmma32(af, bf, acc);
        }
        const float bs = bias[col];
        const float gs = gamma[col] * bn_rs();
        const float bt = beta[col];
        float m = -FLT_MAX;
#pragma unroll
        for (int v = 0; v < 8; ++v) {
            int row = mt * 16 + v + hsel * 8;
            if (row < myCnt) {
                float y = fmaxf(acc[v] + bs, 0.0f) * gs + bt;
                m = fmaxf(m, y);
            }
        }
        m = fmaxf(m, __shfl_xor(m, 16, 32));   // combine hsel halves (same col)
        if (hsel == 0) atomicMaxF32(&colmax[col], m);
    }
}

// ---------------------------------------------------------------------------
// Fused PointConv: one block (256 threads = 8 waves) per center.
// Gathers up to 64 neighbor feature rows into LDS, runs 3 WMMA MLP layers,
// and writes the masked per-channel max.
// ---------------------------------------------------------------------------
__global__ void __launch_bounds__(256) pointconv_kernel(
        const float* __restrict__ pos, const float* __restrict__ xin,
        const float* __restrict__ centers, const int* __restrict__ nidx,
        const int* __restrict__ cnt,
        const _Float16* __restrict__ Wt0, const float* __restrict__ b0,
        const float* __restrict__ g0, const float* __restrict__ be0,
        const _Float16* __restrict__ Wt1, const float* __restrict__ b1,
        const float* __restrict__ g1, const float* __restrict__ be1,
        const _Float16* __restrict__ Wt2, const float* __restrict__ b2,
        const float* __restrict__ g2, const float* __restrict__ be2,
        float* __restrict__ xout,
        int Nsrc, int M, int Cf, int cin0Pad, int c0, int c1, int c2) {
    __shared__ __align__(16) _Float16 bufA[64 * 264];
    __shared__ __align__(16) _Float16 bufB[64 * 264];
    __shared__ float colmax[256];

    const int blk = blockIdx.x;
    const int b = blk / M;
    const int cbase = blk;  // == b*M + m
    const int myCnt = cnt[cbase];
    const float cx = centers[cbase * 3 + 0];
    const float cy = centers[cbase * 3 + 1];
    const float cz = centers[cbase * 3 + 2];

    // Build the [64 x cin0Pad] feature tile: [x_in(j), pos(j)-center], zero pad.
    const int total = 64 * cin0Pad;
    for (int idx = threadIdx.x; idx < total; idx += 256) {
        int r = idx / cin0Pad;
        int c = idx - r * cin0Pad;
        float v = 0.0f;
        if (r < myCnt) {
            int j = nidx[cbase * 64 + r];
            int gi = b * Nsrc + j;
            if (c < Cf) {
                v = xin[gi * Cf + c];
            } else if (c < Cf + 3) {
                int d = c - Cf;
                float pv = pos[gi * 3 + d];
                v = pv - (d == 0 ? cx : (d == 1 ? cy : cz));
            }
        }
        bufA[r * 264 + c] = (_Float16)v;
    }
    if (threadIdx.x < 256) colmax[threadIdx.x] = -FLT_MAX;
    __syncthreads();

    pc_layer(bufA, bufB, Wt0, b0, g0, be0, cin0Pad, c0);
    __syncthreads();
    pc_layer(bufB, bufA, Wt1, b1, g1, be1, c0, c1);
    __syncthreads();
    pc_layer_final(bufA, Wt2, b2, g2, be2, colmax, myCnt, c1, c2);
    __syncthreads();

    for (int c = threadIdx.x; c < c2; c += 256)
        xout[cbase * c2 + c] = colmax[c];
}

// ---------------------------------------------------------------------------
// Generic WMMA GEMM with ReLU+BN epilogue for mlp3: Y = bn(relu(X @ W + b)).
// X: [R][cinPad] f16 (rows multiple of 64), Wt: [cout][cinPad] f16.
// Each wave computes a 64x16 output tile (4 M-tiles sharing each B fragment).
// ---------------------------------------------------------------------------
__global__ void __launch_bounds__(256) gemm_mlp_kernel(
        const _Float16* __restrict__ X, const _Float16* __restrict__ Wt,
        const float* __restrict__ bias, const float* __restrict__ gamma,
        const float* __restrict__ beta, _Float16* __restrict__ Y,
        int R, int cinPad, int cout) {
    const int lane = threadIdx.x & 31;
    const int wave = threadIdx.x >> 5;
    const int nlo = lane & 15;
    const int hsel = (lane >> 4) & 1;
    const int rtiles = R >> 6;
    const int g = blockIdx.x * 8 + wave;
    if (g >= rtiles * (cout >> 4)) return;  // whole-wave exit keeps EXEC full
    const int rt = g % rtiles;
    const int ct = g / rtiles;
    const int col = ct * 16 + nlo;

    v8f acc[4] = {};
    for (int kc = 0; kc < cinPad; kc += 32) {
        __builtin_prefetch(Wt + col * cinPad + kc + 32 + hsel * 8, 0, 1);
        v16h bf = load_frag(Wt + col * cinPad + kc + hsel * 8);
        const _Float16* xb = X + (rt * 64 + nlo) * cinPad + kc + hsel * 8;
#pragma unroll
        for (int t = 0; t < 4; ++t)
            acc[t] = wmma32(load_frag(xb + t * 16 * cinPad), bf, acc[t]);
    }
    const float bs = bias[col];
    const float gs = gamma[col] * bn_rs();
    const float bt = beta[col];
#pragma unroll
    for (int t = 0; t < 4; ++t) {
#pragma unroll
        for (int v = 0; v < 8; ++v) {
            int row = rt * 64 + t * 16 + v + hsel * 8;
            float y = fmaxf(acc[t][v] + bs, 0.0f) * gs + bt;
            Y[row * cout + col] = (_Float16)y;
        }
    }
}

// ---------------------------------------------------------------------------
// Small utility kernels
// ---------------------------------------------------------------------------
__global__ void wcvt_kernel(const float* __restrict__ W, _Float16* __restrict__ Wt,
                            int Cin, int Cout, int CinPad, int total) {
    int t = blockIdx.x * 256 + threadIdx.x;
    if (t >= total) return;
    int n = t / CinPad;
    int k = t - n * CinPad;
    Wt[t] = (k < Cin) ? (_Float16)W[k * Cout + n] : (_Float16)0.0f;
}

__global__ void buildg_kernel(const float* __restrict__ x2, const float* __restrict__ cen,
                              _Float16* __restrict__ X, int C, int CinPad, int total) {
    int t = blockIdx.x * 256 + threadIdx.x;
    if (t >= total) return;
    int r = t / CinPad;
    int c = t - r * CinPad;
    float v = 0.0f;
    if (c < C) v = x2[r * C + c];
    else if (c < C + 3) v = cen[r * 3 + (c - C)];
    X[t] = (_Float16)v;
}

__global__ void maxpool_kernel(const _Float16* __restrict__ Y, float* __restrict__ g,
                               int Rows, int C, int total) {
    int t = blockIdx.x * 256 + threadIdx.x;
    if (t >= total) return;
    int b = t / C;
    int c = t - b * C;
    float m = -FLT_MAX;
    for (int r = 0; r < Rows; ++r)
        m = fmaxf(m, (float)Y[(b * Rows + r) * C + c]);
    g[t] = m;
}

__global__ void linear_kernel(const float* __restrict__ X, const float* __restrict__ W,
                              const float* __restrict__ bvec, float* __restrict__ Y,
                              int Cin, int Cout, int doRelu, int total) {
    int t = blockIdx.x * 256 + threadIdx.x;
    if (t >= total) return;
    int r = t / Cout;
    int c = t - r * Cout;
    float acc = bvec[c];
    for (int k = 0; k < Cin; ++k)
        acc += X[r * Cin + k] * W[k * Cout + c];
    if (doRelu) acc = fmaxf(acc, 0.0f);
    Y[t] = acc;
}

// ---------------------------------------------------------------------------
// Host orchestration
// ---------------------------------------------------------------------------
static inline int cdiv(int a, int b) { return (a + b - 1) / b; }

extern "C" void kernel_launch(void* const* d_in, const int* in_sizes, int n_in,
                              void* d_out, int out_size, void* d_ws, size_t ws_size,
                              hipStream_t stream) {
    (void)in_sizes; (void)out_size;
    if (n_in < 43) return;

    // Input layout (setup_inputs dict insertion order):
    // 0: pos [4,4096,3]
    // 1..12 : mlp1 layers {W,b,gamma,beta} x3
    // 13..24: mlp2 layers x3
    // 25..36: mlp3 layers x3
    // 37..42: lin1.W, lin1.b, lin2.W, lin2.b, lin3.W, lin3.b
    const float* pos = (const float*)d_in[0];
    const float *W[9], *bb[9], *gg[9], *be[9];
    int p = 1;
    for (int l = 0; l < 9; ++l) {
        W[l]  = (const float*)d_in[p++];
        bb[l] = (const float*)d_in[p++];
        gg[l] = (const float*)d_in[p++];
        be[l] = (const float*)d_in[p++];
    }
    const float* lw1 = (const float*)d_in[p++]; const float* lb1 = (const float*)d_in[p++];
    const float* lw2 = (const float*)d_in[p++]; const float* lb2 = (const float*)d_in[p++];
    const float* lw3 = (const float*)d_in[p++]; const float* lb3 = (const float*)d_in[p++];

    const int B = 4, N1 = 4096, M1 = 2048, M2 = 512;
    const int cin[9]  = {3, 64, 64,   131, 128, 128,   259, 256, 512};
    const int cinP[9] = {32, 64, 64,  160, 128, 128,   288, 256, 512};
    const int cout[9] = {64, 64, 128, 128, 128, 256,   256, 512, 1024};

    // Workspace carve (256B aligned slabs).
    uintptr_t base = (uintptr_t)d_ws;
    size_t off = 0;
    auto alloc = [&](size_t bytes) -> void* {
        off = (off + 255) & ~(size_t)255;
        void* r = (void*)(base + off);
        off += bytes;
        return r;
    };
    _Float16* Wt[9];
    for (int l = 0; l < 9; ++l)
        Wt[l] = (_Float16*)alloc((size_t)cout[l] * cinP[l] * sizeof(_Float16));
    int*   fps1  = (int*)alloc((size_t)B * M1 * sizeof(int));
    float* p1    = (float*)alloc((size_t)B * M1 * 3 * sizeof(float));
    int*   nidx1 = (int*)alloc((size_t)B * M1 * 64 * sizeof(int));
    int*   cnt1  = (int*)alloc((size_t)B * M1 * sizeof(int));
    float* d2a   = (float*)alloc((size_t)B * M1 * 64 * sizeof(float));
    float* x1    = (float*)alloc((size_t)B * M1 * 128 * sizeof(float));
    int*   fps2  = (int*)alloc((size_t)B * M2 * sizeof(int));
    float* p2    = (float*)alloc((size_t)B * M2 * 3 * sizeof(float));
    int*   nidx2 = (int*)alloc((size_t)B * M2 * 64 * sizeof(int));
    int*   cnt2  = (int*)alloc((size_t)B * M2 * sizeof(int));
    float* d2b   = (float*)alloc((size_t)B * M2 * 64 * sizeof(float));
    float* x2    = (float*)alloc((size_t)B * M2 * 256 * sizeof(float));
    _Float16* gin = (_Float16*)alloc((size_t)B * M2 * 288 * sizeof(_Float16));
    _Float16* y1 = (_Float16*)alloc((size_t)B * M2 * 256 * sizeof(_Float16));
    _Float16* y2 = (_Float16*)alloc((size_t)B * M2 * 512 * sizeof(_Float16));
    _Float16* y3 = (_Float16*)alloc((size_t)B * M2 * 1024 * sizeof(_Float16));
    float* gvec = (float*)alloc((size_t)B * 1024 * sizeof(float));
    float* h1   = (float*)alloc((size_t)B * 512 * sizeof(float));
    float* h2   = (float*)alloc((size_t)B * 256 * sizeof(float));
    if (off > ws_size) return;

    // 0) Convert/transpose all MLP weights to f16 [cout][cinPad].
    for (int l = 0; l < 9; ++l) {
        int total = cout[l] * cinP[l];
        wcvt_kernel<<<cdiv(total, 256), 256, 0, stream>>>(W[l], Wt[l], cin[l], cout[l], cinP[l], total);
    }

    // 1) SA1: FPS -> centers -> radius -> fused PointConv (3->64->64->128, max)
    fps_kernel<<<B, 1024, 0, stream>>>(pos, fps1, N1, M1);
    gather_centers_kernel<<<cdiv(B * M1, 256), 256, 0, stream>>>(pos, fps1, p1, N1, M1, B * M1);
    radius_kernel<<<cdiv(B * M1, 256), 256, 0, stream>>>(pos, p1, nidx1, cnt1, d2a,
                                                         N1, M1, B * M1, 0.2f * 0.2f);
    pointconv_kernel<<<B * M1, 256, 0, stream>>>(
        pos, nullptr, p1, nidx1, cnt1,
        Wt[0], bb[0], gg[0], be[0],
        Wt[1], bb[1], gg[1], be[1],
        Wt[2], bb[2], gg[2], be[2],
        x1, N1, M1, /*Cf=*/0, /*cin0Pad=*/32, 64, 64, 128);

    // 2) SA2 on (x1, p1): FPS -> radius -> fused PointConv (131->128->128->256, max)
    fps_kernel<<<B, 1024, 0, stream>>>(p1, fps2, M1, M2);
    gather_centers_kernel<<<cdiv(B * M2, 256), 256, 0, stream>>>(p1, fps2, p2, M1, M2, B * M2);
    radius_kernel<<<cdiv(B * M2, 256), 256, 0, stream>>>(p1, p2, nidx2, cnt2, d2b,
                                                         M1, M2, B * M2, 0.4f * 0.4f);
    pointconv_kernel<<<B * M2, 256, 0, stream>>>(
        p1, x1, p2, nidx2, cnt2,
        Wt[3], bb[3], gg[3], be[3],
        Wt[4], bb[4], gg[4], be[4],
        Wt[5], bb[5], gg[5], be[5],
        x2, M1, M2, /*Cf=*/128, /*cin0Pad=*/160, 128, 128, 256);

    // 3) mlp3 on concat(x2, p2): 259->256->512->1024 (WMMA GEMM), then global max.
    {
        int total = B * M2 * 288;
        buildg_kernel<<<cdiv(total, 256), 256, 0, stream>>>(x2, p2, gin, 256, 288, total);
    }
    const int R = B * M2;  // 2048 rows
    gemm_mlp_kernel<<<cdiv((R >> 6) * (256 >> 4), 8), 256, 0, stream>>>(gin, Wt[6], bb[6], gg[6], be[6], y1, R, 288, 256);
    gemm_mlp_kernel<<<cdiv((R >> 6) * (512 >> 4), 8), 256, 0, stream>>>(y1, Wt[7], bb[7], gg[7], be[7], y2, R, 256, 512);
    gemm_mlp_kernel<<<cdiv((R >> 6) * (1024 >> 4), 8), 256, 0, stream>>>(y2, Wt[8], bb[8], gg[8], be[8], y3, R, 512, 1024);
    maxpool_kernel<<<cdiv(B * 1024, 256), 256, 0, stream>>>(y3, gvec, M2, 1024, B * 1024);

    // 4) Final dense head (tiny M=4 rows -> plain f32).
    linear_kernel<<<cdiv(B * 512, 256), 256, 0, stream>>>(gvec, lw1, lb1, h1, 1024, 512, 1, B * 512);
    linear_kernel<<<cdiv(B * 256, 256), 256, 0, stream>>>(h1, lw2, lb2, h2, 512, 256, 1, B * 256);
    linear_kernel<<<cdiv(B * 256, 256), 256, 0, stream>>>(h2, lw3, lb3, (float*)d_out, 256, 256, 0, B * 256);
}
